// SetSequenceModel_14010183320118
// MI455X (gfx1250) — matrix operand order repaired and checked by reading, not compiled, not run on hardware
//
#include <hip/hip_runtime.h>
#include <hip/hip_bf16.h>

// ---------------------------------------------------------------------------
// Types for CDNA5 WMMA
// ---------------------------------------------------------------------------
typedef __attribute__((ext_vector_type(16))) __bf16 v16bf;
typedef __attribute__((ext_vector_type(8)))  float  v8f;

// Problem sizes (fixed by the reference)
#define T_SEQ   1024
#define SET_A   32
#define SET_B   8
#define EMB_D   128
#define H1      512
#define G4      2048   // 4*H1
#define H2      256
#define NCLS    100
#define NPAD    112    // 7 WMMA N-tiles
#define X2D     256    // 2*EMB_D

__device__ __forceinline__ unsigned short f2bf(float x) {
    unsigned int u = __float_as_uint(x);
    unsigned int r = u + 0x7FFFu + ((u >> 16) & 1u);   // round-to-nearest-even
    return (unsigned short)(r >> 16);
}

// ---------------------------------------------------------------------------
// WMMA fragment loaders (ISA 7.12.2 layouts, 16-bit A 16x32 / B 32x16)
// Both x (row-major MxK) and weights (row-major NxK, since gemm is A @ W^T)
// are K-contiguous, so every lane issues 8 dword loads per fragment.
// ---------------------------------------------------------------------------
__device__ __forceinline__ v16bf load_a_frag(const unsigned short* A, int lda,
                                             int m0, int k0, int lane) {
    union { unsigned int i[8]; v16bf v; } u;
    const unsigned short* row = A + (m0 + (lane & 15)) * lda;
    int kh = lane >> 4;
#pragma unroll
    for (int r = 0; r < 8; ++r) {
        int k = k0 + ((r >> 2) << 4) + (kh << 3) + ((r & 3) << 1);
        u.i[r] = *reinterpret_cast<const unsigned int*>(row + k);
    }
    return u.v;
}

__device__ __forceinline__ v16bf load_b_frag(const unsigned short* B, int ldb,
                                             int n0, int k0, int lane) {
    union { unsigned int i[8]; v16bf v; } u;
    const unsigned short* row = B + (n0 + (lane & 15)) * ldb;
    int kh = lane >> 4;
#pragma unroll
    for (int r = 0; r < 8; ++r) {
        int k = k0 + ((r >> 2) << 4) + (kh << 3) + ((r & 3) << 1);
        u.i[r] = *reinterpret_cast<const unsigned int*>(row + k);
    }
    return u.v;
}

// ---------------------------------------------------------------------------
// Prep kernels: f32 -> bf16 weight conversion, padded W_out, bias fold, init
// ---------------------------------------------------------------------------
__global__ void k_f32_to_bf16(const float* __restrict__ src,
                              unsigned short* __restrict__ dst, int n) {
    int i = blockIdx.x * blockDim.x + threadIdx.x;
    if (i < n) dst[i] = f2bf(src[i]);
}

__global__ void k_wout_pad(const float* __restrict__ W,
                           unsigned short* __restrict__ dst) {
    int i = blockIdx.x * blockDim.x + threadIdx.x;   // 112*256 exact
    int row = i >> 8, col = i & 255;
    float v = (row < NCLS) ? W[row * H2 + col] : 0.f;
    dst[i] = f2bf(v);
}

__global__ void k_bias_init(const float* __restrict__ bih,
                            const float* __restrict__ bhh,
                            float* __restrict__ bias,
                            float* __restrict__ hbuf,
                            int* __restrict__ bar) {
    int i = blockIdx.x * blockDim.x + threadIdx.x;
    if (i < G4)  bias[i] = bih[i] + bhh[i];
    if (i < 2 * H1) hbuf[i] = 0.f;
    if (i < 4)   bar[i] = 0;
}

// ---------------------------------------------------------------------------
// Pooling: block = one timestep. Stage 2x256 indices in LDS, 256 lanes stream
// coalesced 512B embedding rows; mean over B folded into a single *0.125.
// ---------------------------------------------------------------------------
__global__ void k_pool(const int* __restrict__ diag, const int* __restrict__ proc,
                       const float* __restrict__ emb, unsigned short* __restrict__ xb) {
    __shared__ int idxs[2][SET_A * SET_B];
    int t = blockIdx.x;
    int tid = threadIdx.x;              // 256
    idxs[0][tid] = diag[t * SET_A * SET_B + tid];
    idxs[1][tid] = proc[t * SET_A * SET_B + tid];
    __syncthreads();
    int half = tid >> 7;                // 0 = diag, 1 = proc
    int d = tid & 127;
    float acc = 0.f;
#pragma unroll 4
    for (int s = 0; s < SET_A * SET_B; ++s)
        acc += emb[idxs[half][s] * EMB_D + d];
    xb[t * X2D + half * EMB_D + d] = f2bf(acc * 0.125f);
}

// ---------------------------------------------------------------------------
// GEMM 1: gx[T,2048] = x[T,256] @ W_ih^T + (b_ih+b_hh), bf16 WMMA, f32 out
// grid (64,16), 256 threads = 8 waves, each wave one 16x16 tile
// ---------------------------------------------------------------------------
__global__ void k_gemm_gx(const unsigned short* __restrict__ xb,
                          const unsigned short* __restrict__ Wb,
                          const float* __restrict__ bias,
                          float* __restrict__ gx) {
    int lane = threadIdx.x & 31, wave = threadIdx.x >> 5;
    int m0 = blockIdx.x * 16;
    int n0 = (blockIdx.y * 8 + wave) * 16;
    v8f c = {};
    for (int k = 0; k < X2D; k += 32) {
        v16bf a = load_a_frag(xb, X2D, m0, k, lane);
        v16bf b = load_b_frag(Wb, X2D, n0, k, lane);
        c = __builtin_amdgcn_wmma_f32_16x16x32_bf16(false, a, false, b,
                                                    (short)0, c, false, false);
    }
    int n = n0 + (lane & 15);
    float bs = bias[n];
    int mb = m0 + ((lane >> 4) << 3);
#pragma unroll
    for (int r = 0; r < 8; ++r)
        gx[(mb + r) * G4 + n] = c[r] + bs;
}

// ---------------------------------------------------------------------------
// LSTM recurrence: 16 persistent workgroups, each owns 32 hidden units and
// caches its 128x512 fp32 W_hh slice in LDS (256 KB of the 320 KB WGP LDS).
// Per step only h (2 KB) crosses WGPs. Sync = gfx1250 cluster barrier
// (s_barrier_signal/-wait -3) + device-scope atomic counter fallback.
// ---------------------------------------------------------------------------
__global__ void k_lstm(const float* __restrict__ gx,
                       const float* __restrict__ Whh,
                       float* __restrict__ hbuf,
                       unsigned short* __restrict__ hsb,
                       int* __restrict__ bar) {
    extern __shared__ float smem[];
    float* Wl   = smem;                    // 128*512 f32 = 256 KB
    float* hl   = smem + 128 * H1;         // 512
    float* dots = hl + H1;                 // 128
    const int tid = threadIdx.x;           // 512
    const int blk = blockIdx.x;            // 16

    // Cache the W_hh rows for our 32 hidden units (4 gates each).
    for (int i = tid; i < 128 * H1; i += 512) {
        int r = i >> 9, k = i & 511;
        int grow = (r >> 5) * H1 + blk * 32 + (r & 31);  // gate*512 + j
        Wl[i] = Whh[grow * H1 + k];
    }
    float cst = 0.f;                       // c-state, owned by tid<32
    __syncthreads();

    const int r = tid >> 2;                // 0..127 gate-row
    const int q = tid & 3;                 // k-quarter

    for (int t = 0; t < T_SEQ; ++t) {
        int cur = t & 1;
        hl[tid] = hbuf[cur * H1 + tid];
        __syncthreads();

        // 4 lanes per row: 128-wide fp32 dot from LDS
        const float* wr = Wl + r * H1 + q * 128;
        const float* hp = hl + q * 128;
        float p = 0.f;
#pragma unroll 8
        for (int k = 0; k < 128; ++k) p = fmaf(wr[k], hp[k], p);
        p += __shfl_xor(p, 1, 4);
        p += __shfl_xor(p, 2, 4);
        if (q == 0) dots[r] = p;
        __syncthreads();

        if (tid < 32) {
            int gj = blk * 32 + tid;
            const float* g = gx + t * G4;
            float iv = g[gj]            + dots[tid];
            float fv = g[H1 + gj]       + dots[32 + tid];
            float gv = g[2 * H1 + gj]   + dots[64 + tid];
            float ov = g[3 * H1 + gj]   + dots[96 + tid];
            iv = 1.f / (1.f + __expf(-iv));
            fv = 1.f / (1.f + __expf(-fv));
            ov = 1.f / (1.f + __expf(-ov));
            gv = tanhf(gv);
            cst = fv * cst + iv * gv;
            float h = ov * tanhf(cst);
            hbuf[((t + 1) & 1) * H1 + gj] = h;
            hsb[t * H1 + gj] = f2bf(h);
        }
        __threadfence();
        __syncthreads();
        if (tid == 0)
            __hip_atomic_fetch_add(&bar[0], 1, __ATOMIC_RELEASE,
                                   __HIP_MEMORY_SCOPE_AGENT);
        // gfx1250 cluster barrier (NOP when not dispatched as a cluster)
        __builtin_amdgcn_s_cluster_barrier();
        if (tid == 0) {
            int target = 16 * (t + 1);
            while (__hip_atomic_load(&bar[0], __ATOMIC_ACQUIRE,
                                     __HIP_MEMORY_SCOPE_AGENT) < target)
                __builtin_amdgcn_s_sleep(1);
        }
        __syncthreads();
        __threadfence();
    }
}

// ---------------------------------------------------------------------------
// GEMM 2: h1 = relu(hs @ W1^T + b1) -> bf16. grid (64,2), 8 waves/block.
// ---------------------------------------------------------------------------
__global__ void k_gemm_h1(const unsigned short* __restrict__ hsb,
                          const unsigned short* __restrict__ W1b,
                          const float* __restrict__ b1,
                          unsigned short* __restrict__ h1b) {
    int lane = threadIdx.x & 31, wave = threadIdx.x >> 5;
    int m0 = blockIdx.x * 16;
    int n0 = (blockIdx.y * 8 + wave) * 16;
    v8f c = {};
    for (int k = 0; k < H1; k += 32) {
        v16bf a = load_a_frag(hsb, H1, m0, k, lane);
        v16bf b = load_b_frag(W1b, H1, n0, k, lane);
        c = __builtin_amdgcn_wmma_f32_16x16x32_bf16(false, a, false, b,
                                                    (short)0, c, false, false);
    }
    int n = n0 + (lane & 15);
    float bs = b1[n];
    int mb = m0 + ((lane >> 4) << 3);
#pragma unroll
    for (int r = 0; r < 8; ++r)
        h1b[(mb + r) * H2 + n] = f2bf(fmaxf(c[r] + bs, 0.f));
}

// ---------------------------------------------------------------------------
// GEMM 3: out = h1 @ W_out^T + b_out, N padded 100->112, f32 out.
// grid (64,1), waves 0..6 active (wave-uniform branch, EXEC all-1 for WMMA).
// ---------------------------------------------------------------------------
__global__ void k_gemm_out(const unsigned short* __restrict__ h1b,
                           const unsigned short* __restrict__ Wob,
                           const float* __restrict__ bout,
                           float* __restrict__ out) {
    int lane = threadIdx.x & 31, wave = threadIdx.x >> 5;
    if (wave >= NPAD / 16) return;
    int m0 = blockIdx.x * 16;
    int n0 = wave * 16;
    v8f c = {};
    for (int k = 0; k < H2; k += 32) {
        v16bf a = load_a_frag(h1b, H2, m0, k, lane);
        v16bf b = load_b_frag(Wob, H2, n0, k, lane);
        c = __builtin_amdgcn_wmma_f32_16x16x32_bf16(false, a, false, b,
                                                    (short)0, c, false, false);
    }
    int n = n0 + (lane & 15);
    if (n < NCLS) {
        float bs = bout[n];
        int mb = m0 + ((lane >> 4) << 3);
#pragma unroll
        for (int r = 0; r < 8; ++r)
            out[(mb + r) * NCLS + n] = c[r] + bs;
    }
}

// ---------------------------------------------------------------------------
// Host-side orchestration
// ---------------------------------------------------------------------------
extern "C" void kernel_launch(void* const* d_in, const int* in_sizes, int n_in,
                              void* d_out, int out_size, void* d_ws, size_t ws_size,
                              hipStream_t stream) {
    (void)in_sizes; (void)n_in; (void)out_size; (void)ws_size;
    const int*   diag = (const int*)  d_in[0];
    const int*   proc = (const int*)  d_in[1];
    const float* emb  = (const float*)d_in[2];
    const float* Wih  = (const float*)d_in[3];
    const float* Whh  = (const float*)d_in[4];
    const float* bih  = (const float*)d_in[5];
    const float* bhh  = (const float*)d_in[6];
    const float* W1   = (const float*)d_in[7];
    const float* b1   = (const float*)d_in[8];
    const float* Wout = (const float*)d_in[9];
    const float* bout = (const float*)d_in[10];
    float* out = (float*)d_out;

    char* ws = (char*)d_ws;
    size_t off = 0;
    unsigned short* xb   = (unsigned short*)(ws + off); off += (size_t)T_SEQ * X2D * 2;   // 512 KB
    unsigned short* Wihb = (unsigned short*)(ws + off); off += (size_t)G4 * X2D * 2;      // 1 MB
    float*          gx   = (float*)         (ws + off); off += (size_t)T_SEQ * G4 * 4;    // 8 MB
    float*          bias = (float*)         (ws + off); off += (size_t)G4 * 4;
    float*          hbuf = (float*)         (ws + off); off += (size_t)2 * H1 * 4;
    unsigned short* hsb  = (unsigned short*)(ws + off); off += (size_t)T_SEQ * H1 * 2;    // 1 MB
    unsigned short* W1b  = (unsigned short*)(ws + off); off += (size_t)H2 * H1 * 2;
    unsigned short* h1b  = (unsigned short*)(ws + off); off += (size_t)T_SEQ * H2 * 2;
    unsigned short* Wob  = (unsigned short*)(ws + off); off += (size_t)NPAD * H2 * 2;
    int*            bar  = (int*)           (ws + off); off += 256;

    // Weight prep (bf16) + state init (hbuf/bar re-zeroed every call: deterministic)
    k_f32_to_bf16<<<(G4 * X2D + 255) / 256, 256, 0, stream>>>(Wih, Wihb, G4 * X2D);
    k_f32_to_bf16<<<(H2 * H1 + 255) / 256, 256, 0, stream>>>(W1, W1b, H2 * H1);
    k_wout_pad<<<NPAD, 256, 0, stream>>>(Wout, Wob);
    k_bias_init<<<8, 256, 0, stream>>>(bih, bhh, bias, hbuf, bar);

    // Gather + pool
    k_pool<<<T_SEQ, 256, 0, stream>>>(diag, proc, emb, xb);

    // Input projection GEMM (WMMA bf16)
    k_gemm_gx<<<dim3(T_SEQ / 16, G4 / 128), 256, 0, stream>>>(xb, Wihb, bias, gx);

    // Sequential recurrence: 16 persistent WGs, 258.5 KB LDS each
    const int LSTM_LDS = (128 * H1 + H1 + 128) * (int)sizeof(float);
    hipFuncSetAttribute((const void*)k_lstm,
                        hipFuncAttributeMaxDynamicSharedMemorySize, LSTM_LDS);
    k_lstm<<<16, 512, LSTM_LDS, stream>>>(gx, Whh, hbuf, hsb, bar);

    // MLP head (WMMA bf16)
    k_gemm_h1<<<dim3(T_SEQ / 16, H2 / 128), 256, 0, stream>>>(hsb, W1b, b1, h1b);
    k_gemm_out<<<dim3(T_SEQ / 16, 1), 256, 0, stream>>>(h1b, Wob, bout, out);
}
